// STGCNBlock_38070590111774
// MI455X (gfx1250) — matrix operand order repaired and checked by reading, not compiled, not run on hardware
//
#include <hip/hip_runtime.h>

// ---------------------------------------------------------------------------
// STGCN block for MI455X (gfx1250, wave32, WMMA).
// conv1 (im2col GEMM K=192) -> sparse graph agg + 64x64 proj -> conv2 + LN.
// All GEMMs: v_wmma_f32_16x16x32_f16. Intermediates f16 in d_ws.
// Weights pre-swizzled in LDS into WMMA B-fragment layout (contiguous per
// lane -> ds_load_b128 pairs). Outputs routed through LDS -> uint4 stores.
// ---------------------------------------------------------------------------

typedef __attribute__((ext_vector_type(16))) _Float16 v16h;
typedef __attribute__((ext_vector_type(8)))  _Float16 v8h;
typedef __attribute__((ext_vector_type(8)))  float    v8f;

constexpr int Bn   = 32;
constexpr int Tt   = 12;
constexpr int Nn   = 883;
constexpr int Cc   = 64;   // == Ff
constexpr int Ff   = 64;
constexpr int ROWS = Bn * Nn * Tt;   // 339072, divisible by 32
constexpr int NTT  = Nn * Tt;        // 10596
constexpr int MAXD = 64;             // adjacency row capacity (~16 expected)

// ---- WMMA fragment loads ---------------------------------------------------

// 16-bit A matrix 16x32 (MxK), per CDNA5 ISA §7.12.2.
// Per lane the halves form two contiguous 8-half (16B) chunks -> 2x ds_load_b128.
__device__ __forceinline__ v16h load_A_frag(const _Float16* At, int stride,
                                            int mt, int kc, int lane) {
  const int row = mt * 16 + (lane & 15);
  const int kg  = (lane >> 4) * 8;       // lanes 16-31 shift K by 8
  const _Float16* p = At + row * stride + kc * 32;
  v16h a;
#pragma unroll
  for (int hi = 0; hi < 16; ++hi) {
    const int v = hi >> 1, sub = hi & 1;
    const int k = ((v >= 4) ? 16 : 0) + kg + (v & 3) * 2 + sub;
    a[hi] = p[k];
  }
  return a;
}

// B fragments pre-swizzled in LDS: frag (kc,nt) stores lane-major 16 halves.
__device__ __forceinline__ v16h load_B_frag_sw(const _Float16* Bsw,
                                               int nt_count, int nt, int kc,
                                               int lane) {
  return *(const v16h*)(Bsw + ((kc * nt_count + nt) * 32 + lane) * 16);
}

// Swizzle staging: W is [Krows x 64] fp32; produce frag layout f16.
// i enumerates (frag=kc*4+nt, lane, hi).
__device__ __forceinline__ void stage_B_swizzled(const float* __restrict__ w,
                                                 _Float16* Bsw, int total,
                                                 int tid) {
  for (int i = tid; i < total; i += 256) {
    const int f    = i >> 9;          // frag id
    const int lane = (i >> 4) & 31;
    const int hi   = i & 15;
    const int kc   = f >> 2;
    const int nt   = f & 3;
    const int row  = kc * 32 + ((hi >= 8) ? 16 : 0) + (lane >> 4) * 8 + (hi & 7);
    const int col  = nt * 16 + (lane & 15);
    Bsw[i] = (_Float16)w[row * 64 + col];
  }
}

// ---- Kernel 0: compact dense adjacency row -> (cnt, cols, vals) ------------
__global__ __launch_bounds__(256) void build_adj_kernel(
    const float* __restrict__ adj, int* __restrict__ cnt,
    int* __restrict__ cols, float* __restrict__ vals) {
  const int n = blockIdx.x * blockDim.x + threadIdx.x;
  if (n >= Nn) return;
  int c = 0;
  for (int m = 0; m < Nn; ++m) {
    const float v = adj[n * Nn + m];
    if (v != 0.0f) {
      if (c < MAXD) { cols[n * MAXD + c] = m; vals[n * MAXD + c] = v; }
      ++c;
    }
  }
  cnt[n] = (c > MAXD) ? MAXD : c;
}

// ---- Kernel 1: conv1 (im2col GEMM, K=192) + bias + ReLU --------------------
// x: [B,T,N,C] fp32.  out: [B,N,T,F] f16 (rows = (b,n,t), t inner).
__global__ __launch_bounds__(256) void conv1_kernel(
    const float* __restrict__ x, const float* __restrict__ w,  // w: [192,64]
    const float* __restrict__ bias, _Float16* __restrict__ out) {
  __shared__ __attribute__((aligned(32))) _Float16 WtS[24 * 512]; // 24 KB
  __shared__ __attribute__((aligned(32))) _Float16 At[32 * 208];  // 13 KB
  __shared__ __attribute__((aligned(16))) _Float16 Ot16[32 * 64]; // 4 KB

  const int tid  = threadIdx.x;
  const int row0 = blockIdx.x * 32;

  stage_B_swizzled(w, WtS, 24 * 512, tid);

  {  // im2col: thread owns (row r, 8-channel segment) for 3 temporal taps
    const int r   = tid >> 3;
    const int c0  = (tid & 7) * 8;
    const int gr  = row0 + r;
    const int b   = gr / NTT;
    const int rem = gr % NTT;
    const int n   = rem / Tt;
    const int t   = rem % Tt;
#pragma unroll
    for (int tap = 0; tap < 3; ++tap) {
      const int ts = t + tap - 1;
      v8h hv = {};
      if (ts >= 0 && ts < Tt) {
        const float* src = x + ((b * Tt + ts) * Nn + n) * Cc + c0;
        const float4 v0 = ((const float4*)src)[0];
        const float4 v1 = ((const float4*)src)[1];
        hv[0] = (_Float16)v0.x; hv[1] = (_Float16)v0.y;
        hv[2] = (_Float16)v0.z; hv[3] = (_Float16)v0.w;
        hv[4] = (_Float16)v1.x; hv[5] = (_Float16)v1.y;
        hv[6] = (_Float16)v1.z; hv[7] = (_Float16)v1.w;
      }
      *(v8h*)(At + r * 208 + tap * 64 + c0) = hv;
    }
  }
  __syncthreads();

  const int lane = tid & 31;
  const int wv   = tid >> 5;
  const int mt   = wv >> 2;        // 0..1
  const int nt   = wv & 3;         // 0..3

  const float bv = bias[nt * 16 + (lane & 15)];
  v8f acc;
#pragma unroll
  for (int r = 0; r < 8; ++r) acc[r] = bv;

#pragma unroll
  for (int kc = 0; kc < 6; ++kc) {
    v16h a = load_A_frag(At, 208, mt, kc, lane);
    v16h b = load_B_frag_sw(WtS, 4, nt, kc, lane);
    acc = __builtin_amdgcn_wmma_f32_16x16x32_f16(false, a, false, b,
                                                 (short)0, acc, false, false);
  }

  const int col = nt * 16 + (lane & 15);
#pragma unroll
  for (int r = 0; r < 8; ++r) {
    const int m = r + (lane >> 4) * 8;
    float v = acc[r];
    v = v > 0.0f ? v : 0.0f;
    Ot16[(mt * 16 + m) * 64 + col] = (_Float16)v;
  }
  __syncthreads();

  // contiguous 4 KB tile store: one uint4 per thread
  ((uint4*)(out + (size_t)row0 * Ff))[tid] = ((const uint4*)Ot16)[tid];
}

// ---- Kernel 2: sparse graph aggregation + 64x64 projection (WMMA) ----------
// h1: [B,N,T,F] f16 -> h2: [B,N,T,F] f16
__global__ __launch_bounds__(256) void agg_proj_kernel(
    const _Float16* __restrict__ h1, const int* __restrict__ cnt,
    const int* __restrict__ cols, const float* __restrict__ vals,
    const float* __restrict__ wg, _Float16* __restrict__ out) {
  __shared__ __attribute__((aligned(32))) _Float16 WGtS[8 * 512];  // 8 KB
  __shared__ __attribute__((aligned(32))) _Float16 Gt[32 * 80];    // 5 KB
  __shared__ __attribute__((aligned(16))) _Float16 Ot16[32 * 64];  // 4 KB

  const int tid  = threadIdx.x;
  const int row0 = blockIdx.x * 32;

  stage_B_swizzled(wg, WGtS, 8 * 512, tid);

  {  // 8 threads/row, 8 channels each: fp32 accumulate over CSR neighbors
    const int r   = tid >> 3;
    const int c0  = (tid & 7) * 8;
    const int gr  = row0 + r;
    const int b   = gr / NTT;
    const int rem = gr % NTT;
    const int n   = rem / Tt;
    const int t   = rem % Tt;
    float a[8];
#pragma unroll
    for (int i = 0; i < 8; ++i) a[i] = 0.0f;
    const int deg = cnt[n];
    for (int j = 0; j < deg; ++j) {
      const int   m = cols[n * MAXD + j];
      const float v = vals[n * MAXD + j];
      const v8h sv = *(const v8h*)(h1 + ((b * Nn + m) * Tt + t) * Ff + c0);
#pragma unroll
      for (int i = 0; i < 8; ++i) a[i] += v * (float)sv[i];
    }
    v8h hv;
#pragma unroll
    for (int i = 0; i < 8; ++i) hv[i] = (_Float16)a[i];
    *(v8h*)(Gt + r * 80 + c0) = hv;
  }
  __syncthreads();

  const int lane = tid & 31;
  const int wv   = tid >> 5;
  const int mt   = wv >> 2;
  const int nt   = wv & 3;

  v8f acc;
#pragma unroll
  for (int r = 0; r < 8; ++r) acc[r] = 0.0f;

#pragma unroll
  for (int kc = 0; kc < 2; ++kc) {     // K = 64
    v16h a = load_A_frag(Gt, 80, mt, kc, lane);
    v16h b = load_B_frag_sw(WGtS, 4, nt, kc, lane);
    acc = __builtin_amdgcn_wmma_f32_16x16x32_f16(false, a, false, b,
                                                 (short)0, acc, false, false);
  }

  const int col = nt * 16 + (lane & 15);
#pragma unroll
  for (int r = 0; r < 8; ++r) {
    const int m = r + (lane >> 4) * 8;
    Ot16[(mt * 16 + m) * 64 + col] = (_Float16)acc[r];
  }
  __syncthreads();

  ((uint4*)(out + (size_t)row0 * Ff))[tid] = ((const uint4*)Ot16)[tid];
}

// ---- Kernel 3: conv2 (im2col GEMM) + bias + ReLU + LayerNorm ---------------
// h2: [B,N,T,F] f16 -> out: [B,T,N,F] fp32 (reference output layout)
__global__ __launch_bounds__(256) void conv2_ln_kernel(
    const _Float16* __restrict__ h2, const float* __restrict__ w,  // [192,64]
    const float* __restrict__ bias, const float* __restrict__ gamma,
    const float* __restrict__ beta, float* __restrict__ out) {
  __shared__ __attribute__((aligned(32))) _Float16 WtS[24 * 512];  // 24 KB
  __shared__ __attribute__((aligned(32))) _Float16 At[32 * 208];   // 13 KB
  __shared__ __attribute__((aligned(16))) float    Ot[32 * 68];    // 8.5 KB

  const int tid  = threadIdx.x;
  const int row0 = blockIdx.x * 32;

  stage_B_swizzled(w, WtS, 24 * 512, tid);

  {  // im2col from f16 input: one v8h (16B) load per tap per thread
    const int r   = tid >> 3;
    const int c0  = (tid & 7) * 8;
    const int gr  = row0 + r;
    const int b   = gr / NTT;
    const int rem = gr % NTT;
    const int n   = rem / Tt;
    const int t   = rem % Tt;
#pragma unroll
    for (int tap = 0; tap < 3; ++tap) {
      const int ts = t + tap - 1;
      v8h hv = {};
      if (ts >= 0 && ts < Tt)
        hv = *(const v8h*)(h2 + ((b * Nn + n) * Tt + ts) * Ff + c0);
      *(v8h*)(At + r * 208 + tap * 64 + c0) = hv;
    }
  }
  __syncthreads();

  const int lane = tid & 31;
  const int wv   = tid >> 5;
  const int mt   = wv >> 2;
  const int nt   = wv & 3;

  const float bv = bias[nt * 16 + (lane & 15)];
  v8f acc;
#pragma unroll
  for (int r = 0; r < 8; ++r) acc[r] = bv;

#pragma unroll
  for (int kc = 0; kc < 6; ++kc) {
    v16h a = load_A_frag(At, 208, mt, kc, lane);
    v16h b = load_B_frag_sw(WtS, 4, nt, kc, lane);
    acc = __builtin_amdgcn_wmma_f32_16x16x32_f16(false, a, false, b,
                                                 (short)0, acc, false, false);
  }

  const int col = nt * 16 + (lane & 15);
#pragma unroll
  for (int r = 0; r < 8; ++r) {
    const int m = r + (lane >> 4) * 8;
    float v = acc[r];
    v = v > 0.0f ? v : 0.0f;
    Ot[(mt * 16 + m) * 68 + col] = v;
  }
  __syncthreads();

  // LayerNorm over F=64: 8 threads/row, shfl reduction within 8-lane group.
  {
    const int r   = tid >> 3;
    const int c0  = (tid & 7) * 8;
    const int gr  = row0 + r;
    const int b   = gr / NTT;
    const int rem = gr % NTT;
    const int n   = rem / Tt;
    const int t   = rem % Tt;

    float vbuf[8];
    float s = 0.0f, s2 = 0.0f;
#pragma unroll
    for (int i = 0; i < 8; ++i) {
      const float v = Ot[r * 68 + c0 + i];
      vbuf[i] = v;
      s  += v;
      s2 += v * v;
    }
#pragma unroll
    for (int off = 1; off < 8; off <<= 1) {
      s  += __shfl_xor(s,  off, 32);
      s2 += __shfl_xor(s2, off, 32);
    }
    const float mu  = s * (1.0f / Ff);
    const float var = s2 * (1.0f / Ff) - mu * mu;
    const float inv = rsqrtf(var + 1e-3f);

    float o[8];
#pragma unroll
    for (int i = 0; i < 8; ++i)
      o[i] = (vbuf[i] - mu) * inv * gamma[c0 + i] + beta[c0 + i];

    float4* dp = (float4*)(out + ((size_t)(b * Tt + t) * Nn + n) * Ff + c0);
    dp[0] = make_float4(o[0], o[1], o[2], o[3]);
    dp[1] = make_float4(o[4], o[5], o[6], o[7]);
  }
}

// ---------------------------------------------------------------------------
extern "C" void kernel_launch(void* const* d_in, const int* in_sizes, int n_in,
                              void* d_out, int out_size, void* d_ws, size_t ws_size,
                              hipStream_t stream) {
  const float* x     = (const float*)d_in[0];
  const float* adj   = (const float*)d_in[1];
  const float* k1    = (const float*)d_in[2];  // [3,1,64,64] == [192,64]
  const float* b1    = (const float*)d_in[3];
  const float* wg    = (const float*)d_in[4];  // [64,64]
  const float* k2    = (const float*)d_in[5];  // [192,64]
  const float* b2    = (const float*)d_in[6];
  const float* gamma = (const float*)d_in[7];
  const float* beta  = (const float*)d_in[8];
  float* out = (float*)d_out;

  // Workspace layout (f16 intermediates + compact adjacency), ~87.5 MB.
  char* ws = (char*)d_ws;
  size_t off = 0;
  auto alloc = [&](size_t bytes) {
    void* p = ws + off;
    off = (off + bytes + 255) & ~size_t(255);
    return p;
  };
  _Float16* h1   = (_Float16*)alloc((size_t)ROWS * Ff * sizeof(_Float16));
  _Float16* h2   = (_Float16*)alloc((size_t)ROWS * Ff * sizeof(_Float16));
  int*      cnt  = (int*)     alloc((size_t)Nn * sizeof(int));
  int*      cols = (int*)     alloc((size_t)Nn * MAXD * sizeof(int));
  float*    vals = (float*)   alloc((size_t)Nn * MAXD * sizeof(float));
  (void)ws_size; (void)in_sizes; (void)n_in; (void)out_size;

  const int tiles = ROWS / 32;  // 10596

  build_adj_kernel<<<(Nn + 255) / 256, 256, 0, stream>>>(adj, cnt, cols, vals);
  conv1_kernel<<<tiles, 256, 0, stream>>>(x, k1, b1, h1);
  agg_proj_kernel<<<tiles, 256, 0, stream>>>(h1, cnt, cols, vals, wg, h2);
  conv2_ln_kernel<<<tiles, 256, 0, stream>>>(h2, k2, b2, gamma, beta, out);
}
